// QuadPath_18897856103243
// MI455X (gfx1250) — matrix operand order repaired and checked by reading, not compiled, not run on hardware
//
#include <hip/hip_runtime.h>
#include <hip/hip_bf16.h>
#include <stdint.h>

// ---------------------------------------------------------------------------
// QuadPath fused kernel for MI455X (gfx1250, wave32, WMMA + TDM).
//   quads[b,k] = x[b, idx[k].x] * x[b, idx[k].y]       (gather-multiply)
//   out[b,o]   = sum_k quads[b,k] * W[o,k]             (65536x2048x2048 GEMM)
//
// Precision: bf16x3 split ("hi/lo") on both quads and W with f32 WMMA
// accumulators -> near-fp32 accuracy; extra WMMAs are free (memory bound:
// HBM floor ~34us vs <1us of bf16 matrix math at WMMA rates).
// Data movement: x tile arrives via Tensor Data Mover (tensor_load_to_lds),
// W is pre-split into bf16 hi/lo in B-fragment order with contiguous k (as A) so B loads
// are 32B/lane contiguous bursts served from L2 (W = 16MB, L2-resident),
// addressed as scalar-base + lane offset to minimize VGPR pressure.
// ---------------------------------------------------------------------------

#define B_ROWS   65536
#define IN_DIM   1024
#define NUM_PROD 2048
#define OUT_DIM  2048

#define MT       32          // rows of B handled per block (2 WMMA M-tiles)
#define KC       128         // K chunk per producer/consumer phase
#define THREADS  512         // 16 waves of 32
#define NWAVES   16
#define NTPW     8           // 16-col N tiles per wave (16*8*16 = 2048 cols)
#define QROW_U16 136         // quad-chunk LDS row stride in u16 (272B, odd banks)
#define BLOB_U16 512         // one B-frag blob = 32 lanes x 16 bf16 = 1KB

typedef __attribute__((ext_vector_type(16))) __bf16       v16bf;
typedef __attribute__((ext_vector_type(8)))  float        v8f;
typedef __attribute__((ext_vector_type(4)))  unsigned int v4u;
typedef __attribute__((ext_vector_type(8)))  int          v8i_t;
typedef __attribute__((ext_vector_type(4)))  int          v4i_t;

#if __has_builtin(__builtin_amdgcn_tensor_load_to_lds)
#define USE_TDM_FILL 1
#else
#define USE_TDM_FILL 0
#endif
#if __has_builtin(__builtin_amdgcn_global_load_async_to_lds_b128)
#define USE_ASYNC_FILL 1
#else
#define USE_ASYNC_FILL 0
#endif

__device__ __forceinline__ void wait_async0() {
#if __has_builtin(__builtin_amdgcn_s_wait_asynccnt)
  __builtin_amdgcn_s_wait_asynccnt(0);
#else
  asm volatile("s_wait_asynccnt 0" ::: "memory");
#endif
}
__device__ __forceinline__ void wait_tensor0() {
#if __has_builtin(__builtin_amdgcn_s_wait_tensorcnt)
  __builtin_amdgcn_s_wait_tensorcnt((unsigned short)0);
#else
  asm volatile("s_wait_tensorcnt 0" ::: "memory");
#endif
}

#if USE_TDM_FILL
// 1-D contiguous TDM copy: nelem f32 from gaddr -> LDS offset lds_off.
// D# per CDNA5 ISA ch.8: group0 {count=1 | lds_addr | global_addr | type=2},
// group1 {data_size=4B, tensor_dim0=nelem, tensor_dim1=1, tile_dim0=nelem,
//         tensor_dim0_stride=nelem}; groups 2/3 zero (<=2D tile).
__device__ __forceinline__ void tdm_load_1d(uint64_t gaddr, uint32_t lds_off,
                                            uint32_t nelem) {
  v4u g0;
  g0[0] = 1u;                                        // count=1, user D#
  g0[1] = lds_off;                                   // lds_addr (bytes)
  g0[2] = (uint32_t)(gaddr & 0xFFFFFFFFu);           // global_addr[31:0]
  g0[3] = (uint32_t)((gaddr >> 32) & 0x01FFFFFFu) | (2u << 30);  // [56:32]|type=2
  v8i_t g1;
  g1[0] = 0x20000;                                   // data_size=2 (4B), mask=0
  g1[1] = (int)((nelem & 0xFFFFu) << 16);            // tensor_dim0[15:0] @bit48
  g1[2] = (int)(((nelem >> 16) & 0xFFFFu) | (1u << 16)); // dim0[31:16]|tensor_dim1=1
  g1[3] = (int)((nelem & 0xFFFFu) << 16);            // tile_dim0 @bit112
  g1[4] = 0;                                         // tile_dim1=0, tile_dim2=0
  g1[5] = (int)nelem;                                // tensor_dim0_stride lo32
  g1[6] = 0;
  g1[7] = 0;
  v4i_t gz4 = {0, 0, 0, 0};
  v8i_t gz8 = {0, 0, 0, 0, 0, 0, 0, 0};
  __builtin_amdgcn_tensor_load_to_lds(g0, g1, gz4, gz4, gz8, 0);
}
#endif

__device__ __forceinline__ uint32_t bf16_rn(float f) {
  uint32_t u = __float_as_uint(f);
  return (u + 0x7FFFu + ((u >> 16) & 1u)) >> 16;
}
__device__ __forceinline__ void split_bf16(float v, uint32_t& h, uint32_t& l) {
  h = bf16_rn(v);
  float hf = __uint_as_float(h << 16);
  l = bf16_rn(v - hf);
}

// ---------------------------------------------------------------------------
// pack_w: split W (f32, [OUT_DIM][NUM_PROD]) into hi/lo bf16 arrays stored in
// contiguous-k B-fragment order (as the A reads), so main-kernel B loads are 32B/lane contiguous:
//   blob(ko,no) = 32 lanes x 16 bf16; lane<16: o=no*16+lane,  k=ko*32+[0,16)
//                                     lane>=16: o=no*16+lane-16, k=ko*32+16+[0,16)
// ---------------------------------------------------------------------------
__global__ void pack_w(const float* __restrict__ W,
                       uint16_t* __restrict__ wh, uint16_t* __restrict__ wl) {
  int t = blockIdx.x * blockDim.x + threadIdx.x;     // one thread per 16-elem blob
  int lane = t & 31;
  int no   = (t >> 5) & 127;
  int ko   = t >> 12;                                // 0..63
  int o    = no * 16 + (lane & 15);
  int kb   = ko * 32 + ((lane >> 4) << 4);
  const float* src = W + (size_t)o * NUM_PROD + kb;
  uint16_t* dh = wh + (size_t)t * 16;
  uint16_t* dl = wl + (size_t)t * 16;
#pragma unroll
  for (int i = 0; i < 16; ++i) {
    uint32_t h, l;
    split_bf16(src[i], h, l);
    dh[i] = (uint16_t)h;
    dl[i] = (uint16_t)l;
  }
}

// ---------------------------------------------------------------------------
// quad_gemm: block = MT rows x full 2048 cols, chunked over K.
// ---------------------------------------------------------------------------
__global__ __launch_bounds__(THREADS, 1)
void quad_gemm(const float* __restrict__ x, const int* __restrict__ idx,
               const uint16_t* __restrict__ wh, const uint16_t* __restrict__ wl,
               float* __restrict__ out) {
  __shared__ float    xb[MT * IN_DIM];          // x tile, row-major [r][c]
  __shared__ uint16_t qh[MT * QROW_U16];        // quads chunk hi (A-frag order)
  __shared__ uint16_t ql[MT * QROW_U16];        // quads chunk lo

  const int tid  = threadIdx.x;
  const int lane = tid & 31;
  const int wavS = __builtin_amdgcn_readfirstlane(tid >> 5);  // scalar wave id
  const long bbase = (long)blockIdx.x * MT;

  // ---- fill x tile: 128KB contiguous copy --------------------------------
#if USE_TDM_FILL
  {
    // one 8KB 1-D TDM tile per wave; descriptor operands are scalar
    uint32_t lds_base =
        (uint32_t)(size_t)(__attribute__((address_space(3))) char*)xb;
    uint64_t ga = (uint64_t)(x + bbase * (size_t)IN_DIM) + (uint64_t)wavS * 8192u;
    tdm_load_1d(ga, lds_base + (uint32_t)wavS * 8192u, 2048u);
  }
  wait_tensor0();
#elif USE_ASYNC_FILL
  {
    const float* src = x + bbase * (size_t)IN_DIM;
#pragma unroll 4
    for (int it = 0; it < (MT * IN_DIM) / (THREADS * 4); ++it) {
      int e = (it * THREADS + tid) * 4;
      __builtin_amdgcn_global_load_async_to_lds_b128(
          (__attribute__((address_space(1))) int*)(src + e),
          (__attribute__((address_space(3))) int*)(&xb[e]), 0, 0);
    }
  }
  wait_async0();
#else
  {
    const float4* src = (const float4*)(x + bbase * (size_t)IN_DIM);
    float4* dst = (float4*)xb;
    for (int it = 0; it < (MT * IN_DIM) / (THREADS * 4); ++it)
      dst[it * THREADS + tid] = src[it * THREADS + tid];
  }
#endif
  __syncthreads();

  v8f acc[2][NTPW] = {};   // 128 accumulator VGPRs per lane

  // producer coordinates: 512 threads = 32 rows x 16 k-groups of 8
  const int pr_r  = tid >> 4;
  const int pr_kg = tid & 15;
  const int group = pr_kg >> 2;
  const int sub   = pr_kg & 3;
  const int slot  = sub;                             // k-order 0-7|8-15|16-23|24-31
  const int st16  = pr_r * QROW_U16 + group * 32 + slot * 8;
  const int xrow  = pr_r << 10;                      // row base in xb

  // consumer: loop-invariant divergent offsets (everything else is scalar)
  const int lane16 = lane * 16;                      // B-frag: lane*32 bytes
  const int a16b   = (lane & 15) * QROW_U16 + ((lane >> 4) << 4);  // A-frag base

  for (int chunk = 0; chunk < NUM_PROD / KC; ++chunk) {
    // ---- producer: compute quads chunk, split hi/lo, store A-frag order ----
    {
      int k0 = chunk * KC + pr_kg * 8;
      const int4* ip = (const int4*)idx + (k0 >> 1);   // 2 index pairs per int4
      uint32_t ph[4], pl[4];
#pragma unroll
      for (int q = 0; q < 4; ++q) {
        int4 pr = ip[q];
        float a0 = xb[xrow + pr.x] * xb[xrow + pr.y];
        float a1 = xb[xrow + pr.z] * xb[xrow + pr.w];
        uint32_t h0, l0, h1, l1;
        split_bf16(a0, h0, l0);
        split_bf16(a1, h1, l1);
        ph[q] = h0 | (h1 << 16);
        pl[q] = l0 | (l1 << 16);
      }
      *(uint4*)(&qh[st16]) = make_uint4(ph[0], ph[1], ph[2], ph[3]);
      *(uint4*)(&ql[st16]) = make_uint4(pl[0], pl[1], pl[2], pl[3]);
    }
    __syncthreads();

    // ---- consumer: WMMA over this K chunk ---------------------------------
    for (int ks = 0; ks < KC / 32; ++ks) {
      int ko = chunk * (KC / 32) + ks;
      // scalar B-frag bases for this wave's 8 blobs (1KB each, nt -> imm off)
      const uint16_t* phb = wh + ((size_t)ko * 128 + wavS * NTPW) * BLOB_U16;
      const uint16_t* plb = wl + ((size_t)ko * 128 + wavS * NTPW) * BLOB_U16;

      // A fragments (hi/lo) for both M tiles, shared across all N tiles
      union V16 { uint4 u[2]; v16bf v; };
      V16 ah[2], al[2];
#pragma unroll
      for (int mt = 0; mt < 2; ++mt) {
        int a16 = a16b + mt * 16 * QROW_U16 + ks * 32;
        ah[mt].u[0] = *(const uint4*)(&qh[a16]);
        ah[mt].u[1] = *(const uint4*)(&qh[a16 + 8]);
        al[mt].u[0] = *(const uint4*)(&ql[a16]);
        al[mt].u[1] = *(const uint4*)(&ql[a16 + 8]);
      }

#pragma unroll
      for (int nt = 0; nt < NTPW; ++nt) {
        v16bf bh = *(const v16bf*)(phb + nt * BLOB_U16 + lane16);
        v16bf bl = *(const v16bf*)(plb + nt * BLOB_U16 + lane16);
#pragma unroll
        for (int mt = 0; mt < 2; ++mt) {
          acc[mt][nt] = __builtin_amdgcn_wmma_f32_16x16x32_bf16(
              false, ah[mt].v, false, bh, (short)0, acc[mt][nt], false, false);
          acc[mt][nt] = __builtin_amdgcn_wmma_f32_16x16x32_bf16(
              false, ah[mt].v, false, bl, (short)0, acc[mt][nt], false, false);
          acc[mt][nt] = __builtin_amdgcn_wmma_f32_16x16x32_bf16(
              false, al[mt].v, false, bh, (short)0, acc[mt][nt], false, false);
        }
      }
    }
    __syncthreads();
  }

  // ---- epilogue: C layout (VGPR v, lane l) -> M = v + 8*(l>=16), N = l%16 --
  // 32-bit offsets from a scalar base (512MB output < 4GB)
  const unsigned rowbase = (unsigned)bbase;
#pragma unroll
  for (int mt = 0; mt < 2; ++mt)
#pragma unroll
    for (int nt = 0; nt < NTPW; ++nt) {
      unsigned n = (unsigned)(wavS * 128 + nt * 16 + (lane & 15));
#pragma unroll
      for (int v = 0; v < 8; ++v) {
        unsigned m = (unsigned)(mt * 16 + ((lane >> 4) << 3) + v);
        out[(rowbase + m) * (unsigned)OUT_DIM + n] = acc[mt][nt][v];
      }
    }
}

// ---------------------------------------------------------------------------
extern "C" void kernel_launch(void* const* d_in, const int* in_sizes, int n_in,
                              void* d_out, int out_size, void* d_ws, size_t ws_size,
                              hipStream_t stream) {
  const float* x   = (const float*)d_in[0];   // [65536,1024] f32
  const int*   idx = (const int*)d_in[1];     // [2048,2] i32
  const float* W   = (const float*)d_in[2];   // [2048,2048] f32
  float* out = (float*)d_out;                 // [65536,2048] f32

  // workspace: W split into hi/lo bf16 in WMMA-B fragment order (16 MB total)
  uint16_t* wh = (uint16_t*)d_ws;
  uint16_t* wl = wh + (size_t)OUT_DIM * NUM_PROD;

  (void)in_sizes; (void)n_in; (void)out_size; (void)ws_size;

  pack_w<<<(OUT_DIM / 16) * (NUM_PROD / 32) * 32 / 256, 256, 0, stream>>>(W, wh, wl);
  quad_gemm<<<B_ROWS / MT, THREADS, 0, stream>>>(x, idx, wh, wl, out);
}
